// BaseDependentAttentionLayer_43688407334933
// MI455X (gfx1250) — compile-verified, matrix operand order checked by reading, not run on hardware
//
#include <hip/hip_runtime.h>
#include <hip/hip_bf16.h>

#define NN 10000
#define MPAD 10112            // NN rounded up to multiple of 128 (block row tile)
#define EE 160000
#define DD 512
#define HH 8
#define HD 64

typedef __attribute__((ext_vector_type(16))) __bf16 v16bf;
typedef __attribute__((ext_vector_type(8)))  __bf16 v8bf;
typedef __attribute__((ext_vector_type(8)))  float  v8f;

// ---------- order-preserving float <-> uint encoding for atomicMax ----------
__device__ __forceinline__ unsigned encf(float f) {
    unsigned u = __float_as_uint(f);
    return (u >> 31) ? ~u : (u | 0x80000000u);
}
__device__ __forceinline__ float decf(unsigned u) {
    return (u & 0x80000000u) ? __uint_as_float(u & 0x7FFFFFFFu)
                             : __uint_as_float(~u);
}

// ---------- weight convert + transpose: W[k][n] f32 -> Wt[n][k] bf16 ----------
__global__ __launch_bounds__(256) void wconv_kernel(const float* __restrict__ W,
                                                    __bf16* __restrict__ Wt) {
    int i = blockIdx.x * blockDim.x + threadIdx.x;   // 0 .. 512*512-1
    int n  = i >> 9;
    int kk = i & 511;
    Wt[(size_t)n * DD + kk] = (__bf16)W[(size_t)kk * DD + n];
}

// ---------- f32 -> bf16 copy ----------
__global__ __launch_bounds__(256) void f2bf_kernel(const float* __restrict__ src,
                                                   __bf16* __restrict__ dst) {
    int i = blockIdx.x * blockDim.x + threadIdx.x;
    dst[i] = (__bf16)src[i];
}

// ---------- LayerNorm: one row per block, writes bf16 (GEMM A operand) ----------
__global__ __launch_bounds__(256) void layernorm_kernel(const float* __restrict__ x,
                                                        const float* __restrict__ g,
                                                        const float* __restrict__ b,
                                                        __bf16* __restrict__ xn) {
    __shared__ float s1[256], s2[256];
    int row = blockIdx.x;
    int t = threadIdx.x;
    const float* xr = x + (size_t)row * DD;
    float a0 = xr[t], a1 = xr[t + 256];
    s1[t] = a0 + a1;
    s2[t] = a0 * a0 + a1 * a1;
    __syncthreads();
    for (int off = 128; off > 0; off >>= 1) {
        if (t < off) { s1[t] += s1[t + off]; s2[t] += s2[t + off]; }
        __syncthreads();
    }
    float mu  = s1[0] * (1.0f / DD);
    float var = s2[0] * (1.0f / DD) - mu * mu;
    float inv = rsqrtf(var + 1e-5f);
    xn[(size_t)row * DD + t]       = (__bf16)((a0 - mu) * inv * g[t]       + b[t]);
    xn[(size_t)row * DD + t + 256] = (__bf16)((a1 - mu) * inv * g[t + 256] + b[t + 256]);
}

// ---------- bf16 fragment load (A or B), row-major, 16B-aligned ----------
__device__ __forceinline__ v16bf load_bf_frag(const __bf16* __restrict__ p) {
    v8bf lo = *(const v8bf*)(p);
    v8bf hi = *(const v8bf*)(p + 16);
    return __builtin_shufflevector(lo, hi, 0,1,2,3,4,5,6,7,8,9,10,11,12,13,14,15);
}

// ---------- WMMA GEMM: C[M x 512] = A[M x 512] @ Wt^T + bias (+ residual) ----------
// A is bf16 row-major, readable for MPAD rows (padded workspace); rows >= M feed
// only accumulators that are never stored. Block = 4 waves; wave tile =
// 32 rows x 64 cols (2 A-frags x 4 B-frags, 8 accumulators): the inner loop is
// pure fragment loads + 8 unconditional WMMAs per k-step (no conversions).
template <bool HAS_RES, bool OUT_BF16>
__global__ __launch_bounds__(128) void gemm_wmma_kernel(const __bf16* __restrict__ A,
                                                        const __bf16* __restrict__ Bt,
                                                        const float* __restrict__ bias,
                                                        const float* __restrict__ res,
                                                        float* __restrict__ Cf,
                                                        __bf16* __restrict__ Cb,
                                                        int M) {
    const int Kd = DD;
    int wave = threadIdx.x >> 5;
    int lane = threadIdx.x & 31;
    int row_base = blockIdx.x * 128 + wave * 32;
    if (row_base >= M) return;                 // wave-uniform
    int rloc = lane & 15;
    int hi   = lane >> 4;                      // 0 or 1
    int koff = hi << 3;                        // 0 or 8
    int col_base = blockIdx.y * 64;

    v8f acc0[4] = {v8f{}, v8f{}, v8f{}, v8f{}};
    v8f acc1[4] = {v8f{}, v8f{}, v8f{}, v8f{}};

    const __bf16* arow0 = A + (size_t)(row_base + rloc) * Kd + koff;
    const __bf16* arow1 = A + (size_t)(row_base + 16 + rloc) * Kd + koff;

    for (int kb = 0; kb < Kd; kb += 32) {
        v16bf af0 = load_bf_frag(arow0 + kb);
        v16bf af1 = load_bf_frag(arow1 + kb);  // padded rows: always loadable

#pragma unroll
        for (int t = 0; t < 4; ++t) {
            v16bf bf = load_bf_frag(Bt + (size_t)(col_base + t * 16 + rloc) * Kd + kb + koff);
            acc0[t] = __builtin_amdgcn_wmma_f32_16x16x32_bf16(
                        false, af0, false, bf, (short)0, acc0[t], false, false);
            acc1[t] = __builtin_amdgcn_wmma_f32_16x16x32_bf16(
                        false, af1, false, bf, (short)0, acc1[t], false, false);
        }
    }

    // C/D layout: VGPR v -> row (v + 8*hi), col = lane&15
#pragma unroll
    for (int t = 0; t < 4; ++t) {
        int col = col_base + t * 16 + rloc;
        float bvv = bias[col];
#pragma unroll
        for (int vv = 0; vv < 8; ++vv) {
            int r = row_base + (hi << 3) + vv;
            size_t idx = (size_t)r * DD + col;
            float val = acc0[t][vv] + bvv;
            if (HAS_RES) val += res[idx];
            if (OUT_BF16) Cb[idx] = (__bf16)val; else Cf[idx] = val;
        }
    }
    if (row_base + 16 < M) {                   // wave-uniform store guard only
#pragma unroll
        for (int t = 0; t < 4; ++t) {
            int col = col_base + t * 16 + rloc;
            float bvv = bias[col];
#pragma unroll
            for (int vv = 0; vv < 8; ++vv) {
                int r = row_base + 16 + (hi << 3) + vv;
                size_t idx = (size_t)r * DD + col;
                float val = acc1[t][vv] + bvv;
                if (HAS_RES) val += res[idx];
                if (OUT_BF16) Cb[idx] = (__bf16)val; else Cf[idx] = val;
            }
        }
    }
}

// ---------- init segmax / denom / values ----------
__global__ __launch_bounds__(256) void fill_init_kernel(unsigned* __restrict__ segmax,
                                                        float* __restrict__ denom,
                                                        float* __restrict__ vals,
                                                        int NH, int ND) {
    int i = blockIdx.x * blockDim.x + threadIdx.x;
    if (i < NH) { segmax[i] = 0x007FFFFFu; /* enc(-inf) */ denom[i] = 0.0f; }
    if (i < ND) vals[i] = 0.0f;
}

// ---------- per-edge scores + segment max ----------
__global__ __launch_bounds__(256) void edge_scores_kernel(const float* __restrict__ q,
                                                          const float* __restrict__ k,
                                                          const long long* __restrict__ ei,
                                                          const float* __restrict__ ew,
                                                          float* __restrict__ wsarr,
                                                          unsigned* __restrict__ segmax) {
    int t = blockIdx.x * blockDim.x + threadIdx.x;
    if (t >= EE * HH) return;
    int e = t >> 3, h = t & 7;
    long long o = ei[e];
    long long d = ei[EE + e];
    const float* qr = q + (size_t)o * DD + h * HD;
    const float* kr = k + (size_t)d * DD + h * HD;
    float acc = 0.0f;
#pragma unroll
    for (int j = 0; j < HD; j += 4) {
        float4 a = *(const float4*)(qr + j);
        float4 b = *(const float4*)(kr + j);
        acc += a.x * b.x + a.y * b.y + a.z * b.z + a.w * b.w;
    }
    float wsv = acc * 0.125f * ew[t];   // SCALE = 64^-0.5
    wsarr[t] = wsv;
    atomicMax(&segmax[(size_t)o * HH + h], encf(wsv));
}

// ---------- exp + segment denom ----------
__global__ __launch_bounds__(256) void edge_exp_kernel(const long long* __restrict__ ei,
                                                       float* __restrict__ wsarr,
                                                       const unsigned* __restrict__ segmax,
                                                       float* __restrict__ denom) {
    int t = blockIdx.x * blockDim.x + threadIdx.x;
    if (t >= EE * HH) return;
    int e = t >> 3, h = t & 7;
    long long o = ei[e];
    float m = decf(segmax[(size_t)o * HH + h]);
    if (m < -3.0e38f) m = 0.0f;         // where(isfinite, m, 0)
    float ev = expf(wsarr[t] - m);
    wsarr[t] = ev;
    atomicAdd(&denom[(size_t)o * HH + h], ev);
}

// ---------- att * v scatter-add ----------
__global__ __launch_bounds__(256) void edge_scatter_kernel(const long long* __restrict__ ei,
                                                           const float* __restrict__ wsarr,
                                                           const float* __restrict__ denom,
                                                           const float* __restrict__ v,
                                                           float* __restrict__ vals) {
    int t = blockIdx.x * blockDim.x + threadIdx.x;
    if (t >= EE * HH) return;
    int e = t >> 3, h = t & 7;
    long long o = ei[e];
    long long d = ei[EE + e];
    float att = wsarr[t] / (denom[(size_t)o * HH + h] + 1e-16f);
    const float* vr = v + (size_t)d * DD + h * HD;
    float* dst = vals + (size_t)o * DD + h * HD;
#pragma unroll 8
    for (int j = 0; j < HD; ++j) {
        atomicAdd(&dst[j], att * vr[j]);
    }
}

extern "C" void kernel_launch(void* const* d_in, const int* in_sizes, int n_in,
                              void* d_out, int out_size, void* d_ws, size_t ws_size,
                              hipStream_t stream) {
    (void)in_sizes; (void)n_in; (void)out_size; (void)ws_size;
    const float*     x    = (const float*)d_in[0];
    const long long* ei   = (const long long*)d_in[1];
    const float*     ew   = (const float*)d_in[2];
    const float*     ln_g = (const float*)d_in[3];
    const float*     ln_b = (const float*)d_in[4];
    const float*     Wq   = (const float*)d_in[5];
    const float*     bq   = (const float*)d_in[6];
    const float*     Wk   = (const float*)d_in[7];
    const float*     bk   = (const float*)d_in[8];
    const float*     Wv   = (const float*)d_in[9];
    const float*     bv_i = (const float*)d_in[10];
    const float*     W1   = (const float*)d_in[11];
    const float*     b1   = (const float*)d_in[12];
    const float*     W2   = (const float*)d_in[13];
    const float*     b2   = (const float*)d_in[14];
    float* out = (float*)d_out;

    const size_t ND4   = (size_t)NN * DD * 4;
    const size_t NDP2  = (size_t)MPAD * DD * 2;   // padded bf16 A-side matrices
    const size_t W2B   = (size_t)DD * DD * 2;
    char* ws = (char*)d_ws;
    size_t off = 0;
    auto take = [&](size_t bytes) -> char* {
        char* p = ws + off;
        off += (bytes + 255) & ~(size_t)255;
        return p;
    };
    __bf16*   xnb   = (__bf16*)take(NDP2);   // GEMM A operands (padded, bf16)
    __bf16*   valsb = (__bf16*)take(NDP2);
    __bf16*   t1b   = (__bf16*)take(NDP2);
    float*    vals  = (float*)take(ND4);     // f32 atomic accumulator
    float*    qb    = (float*)take(ND4);
    float*    kb    = (float*)take(ND4);
    float*    vb    = (float*)take(ND4);
    __bf16*   wtq   = (__bf16*)take(W2B);
    __bf16*   wtk   = (__bf16*)take(W2B);
    __bf16*   wtv   = (__bf16*)take(W2B);
    __bf16*   wt1   = (__bf16*)take(W2B);
    __bf16*   wt2   = (__bf16*)take(W2B);
    float*    wsarr = (float*)take((size_t)EE * HH * 4);
    unsigned* segmx = (unsigned*)take((size_t)NN * HH * 4);
    float*    denom = (float*)take((size_t)NN * HH * 4);

    // 1) weights -> bf16 transposed
    wconv_kernel<<<(DD * DD) / 256, 256, 0, stream>>>(Wq, wtq);
    wconv_kernel<<<(DD * DD) / 256, 256, 0, stream>>>(Wk, wtk);
    wconv_kernel<<<(DD * DD) / 256, 256, 0, stream>>>(Wv, wtv);
    wconv_kernel<<<(DD * DD) / 256, 256, 0, stream>>>(W1, wt1);
    wconv_kernel<<<(DD * DD) / 256, 256, 0, stream>>>(W2, wt2);

    // 2) LayerNorm -> bf16
    layernorm_kernel<<<NN, 256, 0, stream>>>(x, ln_g, ln_b, xnb);

    // 3) q/k/v projections (WMMA bf16)
    dim3 ggrid((NN + 127) / 128, DD / 64);
    gemm_wmma_kernel<false,false><<<ggrid, 128, 0, stream>>>(xnb, wtq, bq, nullptr, qb, nullptr, NN);
    gemm_wmma_kernel<false,false><<<ggrid, 128, 0, stream>>>(xnb, wtk, bk, nullptr, kb, nullptr, NN);
    gemm_wmma_kernel<false,false><<<ggrid, 128, 0, stream>>>(xnb, wtv, bv_i, nullptr, vb, nullptr, NN);

    // 4) init segment state + accumulator
    int ND = NN * DD, NH = NN * HH, EH = EE * HH;
    fill_init_kernel<<<(ND + 255) / 256, 256, 0, stream>>>(segmx, denom, vals, NH, ND);

    // 5) edge phase (L2-resident atomics)
    edge_scores_kernel<<<(EH + 255) / 256, 256, 0, stream>>>(qb, kb, ei, ew, wsarr, segmx);
    edge_exp_kernel<<<(EH + 255) / 256, 256, 0, stream>>>(ei, wsarr, segmx, denom);
    edge_scatter_kernel<<<(EH + 255) / 256, 256, 0, stream>>>(ei, wsarr, denom, vb, vals);

    // 6) output MLP + residual (t1 kept in bf16 end-to-end)
    f2bf_kernel<<<ND / 256, 256, 0, stream>>>(vals, valsb);
    gemm_wmma_kernel<false,true ><<<ggrid, 128, 0, stream>>>(valsb, wt1, b1, nullptr, nullptr, t1b, NN);
    gemm_wmma_kernel<true ,false><<<ggrid, 128, 0, stream>>>(t1b, wt2, b2, x, out, nullptr, NN);
}
